// SelfAttention_25168508355316
// MI455X (gfx1250) — matrix operand order repaired and checked by reading, not compiled, not run on hardware
//
#include <hip/hip_runtime.h>

// ---------------------------------------------------------------------------
// Self-attention forward for MI455X (gfx1250, wave32, WMMA bf16 16x16x32,
// TDM tensor_load_to_lds double-buffered GEMM staging).
// x[2,2048,1024] f32, w_qkv[3072,1024], w_proj[1024,1024] -> out[2,2048,1024]
// ---------------------------------------------------------------------------

typedef __bf16 bf16;
typedef __attribute__((ext_vector_type(16))) __bf16 v16bf;
typedef __attribute__((ext_vector_type(8)))  __bf16 v8bf;
typedef __attribute__((ext_vector_type(8)))  float  v8f;
typedef __attribute__((ext_vector_type(4)))  unsigned int v4u;
typedef __attribute__((ext_vector_type(8)))  int v8i;
typedef __attribute__((ext_vector_type(4)))  int v4i;

#define WMMA_BF16(A, B, C) \
  __builtin_amdgcn_wmma_f32_16x16x32_bf16(false, (A), false, (B), (short)0, (C), false, false)

#define DIMSZ     1024
#define SEQ       2048
#define NHEADS    16
#define HDIM      64
#define BATCH     2
#define MROWS     (BATCH * SEQ)      // 4096
#define ATTN_SCALE 0.125f            // 64^-0.5
#define KSTEPS    (DIMSZ / 32)       // 32 K-iterations per GEMM

// ---------------------------------------------------------------------------
// Tensor Data Mover support (gfx1250). Guarded so the host pass and any
// toolchain lacking the builtin fall back to the sync staging path.
// ---------------------------------------------------------------------------
#if defined(__gfx1250__) && __has_builtin(__builtin_amdgcn_tensor_load_to_lds)
#define USE_TDM 1
#else
#define USE_TDM 0
#endif

#if USE_TDM
#define NBUF 2

// Byte offset of a __shared__ pointer within the workgroup LDS allocation.
static __device__ __attribute__((always_inline)) inline unsigned lds_offset(const void* p) {
  return (unsigned)(unsigned long long)(__attribute__((address_space(3))) const char*)p;
}

// D# group 0: count=1 user descriptor, lds_addr, 57-bit global byte address,
// type=2 ("image").  (ISA 08_async_tensor §8.3)
static __device__ __attribute__((always_inline)) inline v4u
tdm_group0(unsigned lds_addr, unsigned long long gaddr) {
  v4u g;
  g[0] = 1u;
  g[1] = lds_addr;
  g[2] = (unsigned)(gaddr & 0xFFFFFFFFu);
  g[3] = (unsigned)((gaddr >> 32) & 0x01FFFFFFu) | (2u << 30);
  return g;
}

// D# group 1 for a 2D tile of 2-byte elements (data_size=1 -> 2B).
// Fields per ISA 08_async_tensor §8.4 (bit-packed across 8 dwords).
static __device__ __attribute__((always_inline)) inline v8i
tdm_group1_2d(unsigned tensor_d0, unsigned tensor_d1,
              unsigned tile_d0, unsigned tile_d1, unsigned long long stride0) {
  v8i g;
  g[0] = (int)(1u << 16);                                              // data_size=1 (2B)
  g[1] = (int)((tensor_d0 & 0xFFFFu) << 16);                           // tensor_dim0 lo
  g[2] = (int)(((tensor_d0 >> 16) & 0xFFFFu) | ((tensor_d1 & 0xFFFFu) << 16));
  g[3] = (int)(((tensor_d1 >> 16) & 0xFFFFu) | ((tile_d0 & 0xFFFFu) << 16));
  g[4] = (int)(tile_d1 & 0xFFFFu);                                     // tile_dim2 = 0
  g[5] = (int)(stride0 & 0xFFFFFFFFu);                                 // dim0 stride lo
  g[6] = (int)((stride0 >> 32) & 0xFFFFu);                             // hi | dim1_stride=0
  g[7] = 0;
  return g;
}

// Issue one 2D tile load: tile_d1 rows x tile_d0 elements (bf16) from a
// row-major tensor with row stride stride0 (elements) into LDS at lds_addr.
static __device__ __attribute__((always_inline)) inline void
tdm_load_tile(const bf16* gptr, unsigned lds_addr,
              unsigned tensor_d0, unsigned tensor_d1,
              unsigned tile_d0, unsigned tile_d1, unsigned stride0) {
  v4u g0 = tdm_group0(lds_addr, (unsigned long long)(const void*)gptr);
  v8i g1 = tdm_group1_2d(tensor_d0, tensor_d1, tile_d0, tile_d1, stride0);
  v4i z4 = {0, 0, 0, 0};
#if __clang_major__ >= 23
  v8i z8 = {0, 0, 0, 0, 0, 0, 0, 0};
  __builtin_amdgcn_tensor_load_to_lds(g0, g1, z4, z4, z8, 0);
#else
  __builtin_amdgcn_tensor_load_to_lds(g0, g1, z4, z4, 0);
#endif
}

#if __has_builtin(__builtin_amdgcn_s_wait_tensorcnt)
#define WAIT_TENSORCNT(n) __builtin_amdgcn_s_wait_tensorcnt(n)
#else
#define WAIT_TENSORCNT(n) asm volatile("s_wait_tensorcnt 0x0" ::: "memory")
#endif

#else  // !USE_TDM
#define NBUF 1
#endif

// Build a 16-element bf16 fragment from two contiguous 8-element halves.
// Matches ISA layouts: A 16x32 (halves at K=kb and K=kb+16), B 32x16
// (halves at K=kb2 and K=kb2+16).
static __device__ inline v16bf make_frag(const bf16* p0, const bf16* p1) {
  v8bf lo = *(const v8bf*)p0;
  v8bf hi = *(const v8bf*)p1;
  v16bf r;
#pragma unroll
  for (int i = 0; i < 8; ++i) { r[i] = lo[i]; r[i + 8] = hi[i]; }
  return r;
}

// ---------------------------------------------------------------------------
// Kernel 0: f32 -> bf16 conversion (grid-stride).
// ---------------------------------------------------------------------------
__global__ void f32_to_bf16_kernel(const float* __restrict__ src,
                                   bf16* __restrict__ dst, int n) {
  int i = blockIdx.x * blockDim.x + threadIdx.x;
  int stride = gridDim.x * blockDim.x;
  for (; i < n; i += stride) dst[i] = (bf16)src[i];
}

// ---------------------------------------------------------------------------
// Kernel 1: QKV GEMM.  C[4096,3072] = Xbf[4096,1024] @ Wqkv[3072,1024]^T
// Block tile 64(M) x 128(N), 256 threads = 8 waves (4 in M x 2 in N).
// TDM double buffering: wave 0 issues next-tile descriptors, waits
// TENSORcnt<=2 (current tile retired), barrier publishes to all waves.
// Epilogue scatters into Q/K/V [B,H,L,hd] bf16; softmax scale folded into Q.
// ---------------------------------------------------------------------------
__global__ __launch_bounds__(256) void qkv_gemm_kernel(
    const bf16* __restrict__ X, const bf16* __restrict__ W,
    bf16* __restrict__ Q, bf16* __restrict__ Kd, bf16* __restrict__ V) {
  __shared__ __align__(16) bf16 As[NBUF][64][32];
  __shared__ __align__(16) bf16 Bs[NBUF][128][32];

  const int t = threadIdx.x;
  const int lane = t & 31;
  const int wave = t >> 5;
  const int wm = wave >> 1;          // 0..3
  const int wn = wave & 1;           // 0..1
  const int m0 = blockIdx.y * 64;
  const int n0 = blockIdx.x * 128;

  const int kb  = (lane >> 4) * 8;   // A-fragment K base
  const int kb2 = (lane >> 4) * 8;   // B-fragment K base

  v8f acc[4] = {};
  int buf = 0;

#if USE_TDM
  if (wave == 0) {
    tdm_load_tile(X + (size_t)m0 * DIMSZ, lds_offset(&As[0][0][0]),
                  DIMSZ, MROWS, 32, 64, DIMSZ);
    tdm_load_tile(W + (size_t)n0 * DIMSZ, lds_offset(&Bs[0][0][0]),
                  DIMSZ, 3 * DIMSZ, 32, 128, DIMSZ);
  }
  for (int it = 0; it < KSTEPS; ++it) {
    if (wave == 0) {
      if (it + 1 < KSTEPS) {
        const int kn = (it + 1) * 32;
        tdm_load_tile(X + (size_t)m0 * DIMSZ + kn, lds_offset(&As[buf ^ 1][0][0]),
                      DIMSZ, MROWS, 32, 64, DIMSZ);
        tdm_load_tile(W + (size_t)n0 * DIMSZ + kn, lds_offset(&Bs[buf ^ 1][0][0]),
                      DIMSZ, 3 * DIMSZ, 32, 128, DIMSZ);
        WAIT_TENSORCNT(2);           // current tile's 2 descriptors retired
      } else {
        WAIT_TENSORCNT(0);
      }
    }
    __syncthreads();
    const bf16* arp = &As[buf][wm * 16 + (lane & 15)][0];
    v16bf afrag = make_frag(arp + kb, arp + kb + 16);
#pragma unroll
    for (int s = 0; s < 4; ++s) {
      const bf16* brp = &Bs[buf][wn * 64 + s * 16 + (lane & 15)][0];
      v16bf bfrag = make_frag(brp + kb2, brp + kb2 + 16);
      acc[s] = WMMA_BF16(afrag, bfrag, acc[s]);
    }
    __syncthreads();
    buf ^= 1;
  }
#else
  const int arow = t >> 2, achk = (t & 3) * 8;
  const int brow = t >> 1, bchk = (t & 1) * 16;
  for (int k0 = 0; k0 < DIMSZ; k0 += 32) {
    *(v8bf*)&As[0][arow][achk] = *(const v8bf*)&X[(m0 + arow) * DIMSZ + k0 + achk];
    *(v8bf*)&Bs[0][brow][bchk]     = *(const v8bf*)&W[(n0 + brow) * DIMSZ + k0 + bchk];
    *(v8bf*)&Bs[0][brow][bchk + 8] = *(const v8bf*)&W[(n0 + brow) * DIMSZ + k0 + bchk + 8];
    if (k0 + 32 < DIMSZ) {
      __builtin_prefetch(&X[(m0 + arow) * DIMSZ + k0 + 32 + achk], 0, 1);
      __builtin_prefetch(&W[(n0 + brow) * DIMSZ + k0 + 32 + bchk], 0, 1);
    }
    __syncthreads();
    const bf16* arp = &As[0][wm * 16 + (lane & 15)][0];
    v16bf afrag = make_frag(arp + kb, arp + kb + 16);
#pragma unroll
    for (int s = 0; s < 4; ++s) {
      const bf16* brp = &Bs[0][wn * 64 + s * 16 + (lane & 15)][0];
      v16bf bfrag = make_frag(brp + kb2, brp + kb2 + 16);
      acc[s] = WMMA_BF16(afrag, bfrag, acc[s]);
    }
    __syncthreads();
  }
#endif

  // Epilogue: C lane layout -> per-head scatter. n -> (sel, head, d).
#pragma unroll
  for (int s = 0; s < 4; ++s) {
    const int n = n0 + wn * 64 + s * 16 + (lane & 15);
    const int sel = n >> 10;
    const int h = (n >> 6) & (NHEADS - 1);
    const int d = n & (HDIM - 1);
    bf16* dst = (sel == 0) ? Q : ((sel == 1) ? Kd : V);
    const float sc = (sel == 0) ? ATTN_SCALE : 1.0f;
#pragma unroll
    for (int r = 0; r < 8; ++r) {
      const int m = m0 + wm * 16 + r + ((lane >> 4) << 3);
      const int b = m >> 11;
      const int l = m & (SEQ - 1);
      dst[(((b * NHEADS + h) * SEQ) + l) * HDIM + d] = (bf16)(acc[s][r] * sc);
    }
  }
}

// ---------------------------------------------------------------------------
// Kernel 2: flash attention.  One block = 4 waves = 64 query rows of one
// (b,h); each wave owns 16 query rows and streams key blocks of 32.
//   S (16x32) via 4 WMMAs (K-fragments straight from L2-resident K),
//   online softmax with 16-lane shfl_xor row reductions,
//   P -> wave-private LDS -> A-fragment, O += P @ V via LDS-transposed V.
// ---------------------------------------------------------------------------
__global__ __launch_bounds__(128) void attn_kernel(
    const bf16* __restrict__ Q, const bf16* __restrict__ Kd,
    const bf16* __restrict__ V, bf16* __restrict__ AO) {
  __shared__ __align__(16) bf16 VT[64][32];        // [hd][key] transposed V tile
  __shared__ __align__(16) bf16 Ps[4][16][32];     // per-wave P staging

  const int t = threadIdx.x;
  const int lane = t & 31;
  const int wave = t >> 5;
  const int bh = blockIdx.y;                       // b*16 + h
  const int b = bh >> 4, h = bh & 15;
  const int q0 = blockIdx.x * 64 + wave * 16;

  const bf16* qbase = Q  + (size_t)bh * SEQ * HDIM;
  const bf16* kbase = Kd + (size_t)bh * SEQ * HDIM;
  const bf16* vbase = V  + (size_t)bh * SEQ * HDIM;

  // Persistent Q fragments: 16 rows x 64 hd = 2 K-steps of 32.
  const int kb = (lane >> 4) * 8;
  const int qrow = q0 + (lane & 15);
  v16bf qf[2];
#pragma unroll
  for (int s = 0; s < 2; ++s) {
    const bf16* p = qbase + (size_t)qrow * HDIM + s * 32 + kb;
    qf[s] = make_frag(p, p + 16);
  }

  float m_r[8], l_r[8];
  v8f o[4] = {};
#pragma unroll
  for (int r = 0; r < 8; ++r) { m_r[r] = -3.0e38f; l_r[r] = 0.0f; }

  const int vkey = t >> 2;            // 0..31 key within tile
  const int vchk = (t & 3) * 16;      // hd chunk base

  for (int kt = 0; kt < SEQ; kt += 32) {
    // Stage V tile transposed into LDS: VT[hd][key].
    {
      const bf16* vp = vbase + (size_t)(kt + vkey) * HDIM + vchk;
      v8bf x0 = *(const v8bf*)vp;
      v8bf x1 = *(const v8bf*)(vp + 8);
#pragma unroll
      for (int i = 0; i < 8; ++i) {
        VT[vchk + i][vkey]     = x0[i];
        VT[vchk + 8 + i][vkey] = x1[i];
      }
    }
    __syncthreads();

    // S = Q @ K^T for 32 keys -> two 16x16 C tiles.
    v8f s0 = {}, s1 = {};
    const int kb2 = (lane >> 4) * 8;
#pragma unroll
    for (int st = 0; st < 2; ++st) {
      const bf16* kp0 = kbase + (size_t)(kt + (lane & 15)) * HDIM + st * 32 + kb2;
      v16bf b0 = make_frag(kp0, kp0 + 16);
      s0 = WMMA_BF16(qf[st], b0, s0);
      const bf16* kp1 = kbase + (size_t)(kt + 16 + (lane & 15)) * HDIM + st * 32 + kb2;
      v16bf b1 = make_frag(kp1, kp1 + 16);
      s1 = WMMA_BF16(qf[st], b1, s1);
    }

    // Online softmax. Row r of the C tile lives in VGPR slot r across a
    // 16-lane half -> xor-reduce with masks 1..8 stays within the half.
#pragma unroll
    for (int r = 0; r < 8; ++r) {
      float mx = fmaxf(s0[r], s1[r]);
#pragma unroll
      for (int off = 1; off < 16; off <<= 1) mx = fmaxf(mx, __shfl_xor(mx, off, 32));
      const float mn = fmaxf(m_r[r], mx);
      const float corr = __expf(m_r[r] - mn);
      const float p0 = __expf(s0[r] - mn);
      const float p1 = __expf(s1[r] - mn);
      float rs = p0 + p1;
#pragma unroll
      for (int off = 1; off < 16; off <<= 1) rs += __shfl_xor(rs, off, 32);
      l_r[r] = l_r[r] * corr + rs;
      m_r[r] = mn;
      s0[r] = p0;
      s1[r] = p1;
#pragma unroll
      for (int sub = 0; sub < 4; ++sub) o[sub][r] *= corr;
    }

    // C layout -> A layout via wave-private LDS (wave32: in-order DS).
#pragma unroll
    for (int r = 0; r < 8; ++r) {
      const int prow = r + ((lane >> 4) << 3);
      const int pcol = lane & 15;
      Ps[wave][prow][pcol]      = (bf16)s0[r];
      Ps[wave][prow][pcol + 16] = (bf16)s1[r];
    }
    const bf16* ap = &Ps[wave][lane & 15][0];
    v16bf pfrag = make_frag(ap + kb, ap + kb + 16);

    // O += P @ Vtile  (4 hd sub-tiles, K = 32 keys).
#pragma unroll
    for (int sub = 0; sub < 4; ++sub) {
      const bf16* vp2 = &VT[sub * 16 + (lane & 15)][0] + kb2;
      v16bf bv = make_frag(vp2, vp2 + 16);
      o[sub] = WMMA_BF16(pfrag, bv, o[sub]);
    }
    __syncthreads();   // protect VT before next tile overwrites it
  }

  // Epilogue: normalize and store to AO[b*SEQ + q][h*64 + d] (bf16).
#pragma unroll
  for (int r = 0; r < 8; ++r) {
    const float inv = 1.0f / l_r[r];
    const int qr = q0 + r + ((lane >> 4) << 3);
#pragma unroll
    for (int sub = 0; sub < 4; ++sub) {
      const int d = sub * 16 + (lane & 15);
      AO[((size_t)(b * SEQ + qr)) * DIMSZ + h * HDIM + d] = (bf16)(o[sub][r] * inv);
    }
  }
}

// ---------------------------------------------------------------------------
// Kernel 3: output projection.  Y[4096,1024] f32 = AO @ Wproj^T.
// Same TDM double-buffered tiling as kernel 1, plain f32 epilogue.
// ---------------------------------------------------------------------------
__global__ __launch_bounds__(256) void proj_gemm_kernel(
    const bf16* __restrict__ X, const bf16* __restrict__ W,
    float* __restrict__ Y) {
  __shared__ __align__(16) bf16 As[NBUF][64][32];
  __shared__ __align__(16) bf16 Bs[NBUF][128][32];

  const int t = threadIdx.x;
  const int lane = t & 31;
  const int wave = t >> 5;
  const int wm = wave >> 1;
  const int wn = wave & 1;
  const int m0 = blockIdx.y * 64;
  const int n0 = blockIdx.x * 128;

  const int kb  = (lane >> 4) * 8;
  const int kb2 = (lane >> 4) * 8;

  v8f acc[4] = {};
  int buf = 0;

#if USE_TDM
  if (wave == 0) {
    tdm_load_tile(X + (size_t)m0 * DIMSZ, lds_offset(&As[0][0][0]),
                  DIMSZ, MROWS, 32, 64, DIMSZ);
    tdm_load_tile(W + (size_t)n0 * DIMSZ, lds_offset(&Bs[0][0][0]),
                  DIMSZ, DIMSZ, 32, 128, DIMSZ);
  }
  for (int it = 0; it < KSTEPS; ++it) {
    if (wave == 0) {
      if (it + 1 < KSTEPS) {
        const int kn = (it + 1) * 32;
        tdm_load_tile(X + (size_t)m0 * DIMSZ + kn, lds_offset(&As[buf ^ 1][0][0]),
                      DIMSZ, MROWS, 32, 64, DIMSZ);
        tdm_load_tile(W + (size_t)n0 * DIMSZ + kn, lds_offset(&Bs[buf ^ 1][0][0]),
                      DIMSZ, DIMSZ, 32, 128, DIMSZ);
        WAIT_TENSORCNT(2);
      } else {
        WAIT_TENSORCNT(0);
      }
    }
    __syncthreads();
    const bf16* arp = &As[buf][wm * 16 + (lane & 15)][0];
    v16bf afrag = make_frag(arp + kb, arp + kb + 16);
#pragma unroll
    for (int s = 0; s < 4; ++s) {
      const bf16* brp = &Bs[buf][wn * 64 + s * 16 + (lane & 15)][0];
      v16bf bfrag = make_frag(brp + kb2, brp + kb2 + 16);
      acc[s] = WMMA_BF16(afrag, bfrag, acc[s]);
    }
    __syncthreads();
    buf ^= 1;
  }
#else
  const int arow = t >> 2, achk = (t & 3) * 8;
  const int brow = t >> 1, bchk = (t & 1) * 16;
  for (int k0 = 0; k0 < DIMSZ; k0 += 32) {
    *(v8bf*)&As[0][arow][achk] = *(const v8bf*)&X[(m0 + arow) * DIMSZ + k0 + achk];
    *(v8bf*)&Bs[0][brow][bchk]     = *(const v8bf*)&W[(n0 + brow) * DIMSZ + k0 + bchk];
    *(v8bf*)&Bs[0][brow][bchk + 8] = *(const v8bf*)&W[(n0 + brow) * DIMSZ + k0 + bchk + 8];
    if (k0 + 32 < DIMSZ) {
      __builtin_prefetch(&X[(m0 + arow) * DIMSZ + k0 + 32 + achk], 0, 1);
      __builtin_prefetch(&W[(n0 + brow) * DIMSZ + k0 + 32 + bchk], 0, 1);
    }
    __syncthreads();
    const bf16* arp = &As[0][wm * 16 + (lane & 15)][0];
    v16bf afrag = make_frag(arp + kb, arp + kb + 16);
#pragma unroll
    for (int s = 0; s < 4; ++s) {
      const bf16* brp = &Bs[0][wn * 64 + s * 16 + (lane & 15)][0];
      v16bf bfrag = make_frag(brp + kb2, brp + kb2 + 16);
      acc[s] = WMMA_BF16(afrag, bfrag, acc[s]);
    }
    __syncthreads();
  }
#endif

#pragma unroll
  for (int s = 0; s < 4; ++s) {
    const int n = n0 + wn * 64 + s * 16 + (lane & 15);
#pragma unroll
    for (int r = 0; r < 8; ++r) {
      const int m = m0 + wm * 16 + r + ((lane >> 4) << 3);
      Y[(size_t)m * DIMSZ + n] = acc[s][r];
    }
  }
}

// ---------------------------------------------------------------------------
// Launch: convert -> QKV GEMM -> attention -> projection.
// Workspace layout (bytes):
//   [0,8M)    x_bf      4096x1024 bf16
//   [8M,14M)  wqkv_bf   3072x1024 bf16
//   [14M,16M) wproj_bf  1024x1024 bf16
//   [16M,24M) q_bf      [2,16,2048,64] bf16 (pre-scaled)
//   [24M,32M) k_bf
//   [32M,40M) v_bf
//   [40M,48M) ao_bf     4096x1024 bf16
// ---------------------------------------------------------------------------
extern "C" void kernel_launch(void* const* d_in, const int* in_sizes, int n_in,
                              void* d_out, int out_size, void* d_ws, size_t ws_size,
                              hipStream_t stream) {
  const float* x      = (const float*)d_in[0];
  const float* w_qkv  = (const float*)d_in[1];
  const float* w_proj = (const float*)d_in[2];
  float* out = (float*)d_out;

  char* ws = (char*)d_ws;
  bf16* x_bf     = (bf16*)(ws);
  bf16* wqkv_bf  = (bf16*)(ws + 8388608);
  bf16* wproj_bf = (bf16*)(ws + 14680064);
  bf16* q_bf     = (bf16*)(ws + 16777216);
  bf16* k_bf     = (bf16*)(ws + 25165824);
  bf16* v_bf     = (bf16*)(ws + 33554432);
  bf16* ao_bf    = (bf16*)(ws + 41943040);

  f32_to_bf16_kernel<<<dim3(2048), 256, 0, stream>>>(x,      x_bf,     MROWS * DIMSZ);
  f32_to_bf16_kernel<<<dim3(1536), 256, 0, stream>>>(w_qkv,  wqkv_bf,  3 * DIMSZ * DIMSZ);
  f32_to_bf16_kernel<<<dim3(512),  256, 0, stream>>>(w_proj, wproj_bf, DIMSZ * DIMSZ);

  qkv_gemm_kernel<<<dim3(24, 64), 256, 0, stream>>>(x_bf, wqkv_bf, q_bf, k_bf, v_bf);
  attn_kernel<<<dim3(32, 32), 128, 0, stream>>>(q_bf, k_bf, v_bf, ao_bf);
  proj_gemm_kernel<<<dim3(8, 64), 256, 0, stream>>>(ao_bf, wproj_bf, out);
}
